// LSTMForecaster_36885179138477
// MI455X (gfx1250) — matrix operand order repaired and checked by reading, not compiled, not run on hardware
//
#include <hip/hip_runtime.h>
#include <hip/hip_bf16.h>

typedef __attribute__((ext_vector_type(16))) _Float16 v16h;
typedef __attribute__((ext_vector_type(8)))  _Float16 v8h;
typedef __attribute__((ext_vector_type(8)))  float    v8f;

#define T_LEN 512
#define GS 260     // gates LDS row stride (floats): 4-bank row skew, conflict-free
#define HS 72      // hidden LDS row stride (halves): rows map to distinct bank groups

// ---- convert the three 256x64 weight matrices to f16 (row-major kept) ----
__global__ void prep_weights_f16(const float* __restrict__ Whh0,
                                 const float* __restrict__ Wih1,
                                 const float* __restrict__ Whh1,
                                 _Float16* __restrict__ w16) {
    int i = blockIdx.x * blockDim.x + threadIdx.x;
    if (i < 256 * 64) {
        w16[i]           = (_Float16)Whh0[i];
        w16[16384 + i]   = (_Float16)Wih1[i];
        w16[32768 + i]   = (_Float16)Whh1[i];
    }
}

static __device__ __forceinline__ v8f wmma_f16(v16h a, v16h b, v8f c) {
    return __builtin_amdgcn_wmma_f32_16x16x32_f16(
        /*neg_a=*/false, a, /*neg_b=*/false, b,
        /*c_mod=*/(short)0, c, /*reuse_a=*/false, /*reuse_b=*/false);
}

static __device__ __forceinline__ v16h cat16(v8h lo, v8h hi) {
    return __builtin_shufflevector(lo, hi, 0,1,2,3,4,5,6,7,8,9,10,11,12,13,14,15);
}

// A-matrix 16x32 f16 fragment from LDS hidden buffer (stride HS halves).
// lane<16: row=lane, K = kb*32 + {0..7, 16..23}; lane>=16: row=lane-16, K = kb*32 + {8..15, 24..31}
static __device__ __forceinline__ v16h load_a_frag(const _Float16* hbuf, int lane, int kb) {
    const int row = lane & 15;
    const int sub = lane >> 4;
    const _Float16* base = hbuf + row * HS + kb * 32 + sub * 8;
    v8h lo = *(const v8h*)(base);
    v8h hi = *(const v8h*)(base + 16);
    return cat16(lo, hi);
}

// B-matrix 32x16 f16 fragment from global f16 weights (row-major 256x64, W[n][k]).
// col n = tile base + (lane&15); K = kb*32 + sub*8 + {0..7, 16..23}
static __device__ __forceinline__ v16h load_b_frag(const _Float16* W, int n, int sub, int kb) {
    const _Float16* base = W + n * 64 + kb * 32 + sub * 8;
    v8h lo = *(const v8h*)(base);
    v8h hi = *(const v8h*)(base + 16);
    return cat16(lo, hi);
}

// ---- branch-free fast activations on the gfx1250 TRANS pipe ----
#if __has_builtin(__builtin_amdgcn_tanhf)
static __device__ __forceinline__ float fast_tanh(float x) {
    return __builtin_amdgcn_tanhf(x);              // v_tanh_f32 (CDNA5 hardware trans op)
}
static __device__ __forceinline__ float fast_sigm(float x) {
    // sigmoid(x) = 0.5*tanh(x/2) + 0.5  -> 1 trans + 2 VALU, no branches
    return fmaf(0.5f, __builtin_amdgcn_tanhf(0.5f * x), 0.5f);
}
#else
static __device__ __forceinline__ float fast_tanh(float x) {
    // tanh(x) = 1 - 2/(2^(2x*log2e) + 1); v_exp_f32 is full-range in HW, no reduction
    float e = __builtin_amdgcn_exp2f(x * 2.8853900817779268f);
    return fmaf(-2.0f, __builtin_amdgcn_rcpf(e + 1.0f), 1.0f);
}
static __device__ __forceinline__ float fast_sigm(float x) {
    float e = __builtin_amdgcn_exp2f(-x * 1.4426950408889634f);
    return __builtin_amdgcn_rcpf(1.0f + e);
}
#endif

__global__ __launch_bounds__(256) void lstm2_fused(
    const float* __restrict__ x,      // (B, T, 1)
    const float* __restrict__ Wih0,   // (256, 1)
    const float* __restrict__ bih0, const float* __restrict__ bhh0,
    const float* __restrict__ bih1, const float* __restrict__ bhh1,
    const float* __restrict__ Wfc,    // (1, 64)
    const float* __restrict__ bfc,    // (1,)
    const _Float16* __restrict__ w16, // [Whh0 | Wih1 | Whh1] f16, each 256x64 row-major
    float* __restrict__ out)          // (B, 1)
{
    __shared__ float    xs[16 * T_LEN];    // batch-tile of x, 32 KB
    __shared__ float    gates[16 * GS];    // raw gates (one layer at a time)
    __shared__ _Float16 h0s[16 * HS];      // layer0 hidden (f16)
    __shared__ _Float16 h1s[16 * HS];      // layer1 hidden (f16)
    __shared__ float    hf[16 * 64];       // final layer1 hidden (f32)

    const int tid   = threadIdx.x;
    const int lane  = tid & 31;
    const int wave  = tid >> 5;
    const int bbase = blockIdx.x * 16;
    const int colL  = lane & 15;
    const int sub   = lane >> 4;

    // init hidden states to zero; preload x tile (coalesced along t)
    for (int i = tid; i < 16 * HS; i += 256) { h0s[i] = (_Float16)0.f; h1s[i] = (_Float16)0.f; }
    for (int i = tid; i < 16 * T_LEN; i += 256) {
        int r = i >> 9, t = i & (T_LEN - 1);
        xs[i] = x[(bbase + r) * T_LEN + t];
    }

    // this wave's two gate-column tiles
    const int n0 = wave * 32 + colL;
    const int n1 = n0 + 16;

    // per-lane layer0 x-projection scalars and fused biases
    const float w0a = Wih0[n0],              w0b = Wih0[n1];
    const float c0a = bih0[n0] + bhh0[n0],   c0b = bih0[n1] + bhh0[n1];
    const float c1a = bih1[n0] + bhh1[n0],   c1b = bih1[n1] + bhh1[n1];

    const _Float16* Whh0h = w16;
    const _Float16* Wih1h = w16 + 16384;
    const _Float16* Whh1h = w16 + 32768;

    // weight-stationary B fragments: 12 x v16h, live in VGPRs for all 512 steps
    v16h Bhh0_a0 = load_b_frag(Whh0h, n0, sub, 0), Bhh0_a1 = load_b_frag(Whh0h, n0, sub, 1);
    v16h Bhh0_b0 = load_b_frag(Whh0h, n1, sub, 0), Bhh0_b1 = load_b_frag(Whh0h, n1, sub, 1);
    v16h Bih1_a0 = load_b_frag(Wih1h, n0, sub, 0), Bih1_a1 = load_b_frag(Wih1h, n0, sub, 1);
    v16h Bih1_b0 = load_b_frag(Wih1h, n1, sub, 0), Bih1_b1 = load_b_frag(Wih1h, n1, sub, 1);
    v16h Bhh1_a0 = load_b_frag(Whh1h, n0, sub, 0), Bhh1_a1 = load_b_frag(Whh1h, n0, sub, 1);
    v16h Bhh1_b0 = load_b_frag(Whh1h, n1, sub, 0), Bhh1_b1 = load_b_frag(Whh1h, n1, sub, 1);

    // persistent f32 cell state: thread owns 4 rows of one hidden unit, both layers
    float cell0[4] = {0.f, 0.f, 0.f, 0.f};
    float cell1[4] = {0.f, 0.f, 0.f, 0.f};
    const int ej  = tid & 63;          // hidden unit
    const int er0 = (tid >> 6) << 2;   // first of 4 rows

    __syncthreads();

    for (int t = 0; t < T_LEN; ++t) {
        // ---------------- layer 0: gates = x_proj + h0 @ Whh0^T ----------------
        v8f acc0, acc1;
        #pragma unroll
        for (int v = 0; v < 8; ++v) {
            float xv = xs[(v + sub * 8) * T_LEN + t];   // broadcast LDS read
            acc0[v] = fmaf(xv, w0a, c0a);
            acc1[v] = fmaf(xv, w0b, c0b);
        }
        {
            v16h A0 = load_a_frag(h0s, lane, 0);
            v16h A1 = load_a_frag(h0s, lane, 1);
            acc0 = wmma_f16(A0, Bhh0_a0, acc0);
            acc0 = wmma_f16(A1, Bhh0_a1, acc0);
            acc1 = wmma_f16(A0, Bhh0_b0, acc1);
            acc1 = wmma_f16(A1, Bhh0_b1, acc1);
        }
        #pragma unroll
        for (int v = 0; v < 8; ++v) {
            gates[(v + sub * 8) * GS + n0] = acc0[v];
            gates[(v + sub * 8) * GS + n1] = acc1[v];
        }
        __syncthreads();

        // layer 0 elementwise LSTM cell (branch-free activations)
        #pragma unroll
        for (int q = 0; q < 4; ++q) {
            const int r = er0 + q;
            const float* g = gates + r * GS;
            float ig = fast_sigm(g[ej]);
            float fg = fast_sigm(g[64 + ej]);
            float gg = fast_tanh(g[128 + ej]);
            float og = fast_sigm(g[192 + ej]);
            float c  = fmaf(fg, cell0[q], ig * gg);
            cell0[q] = c;
            h0s[r * HS + ej] = (_Float16)(og * fast_tanh(c));
        }
        __syncthreads();

        // ------------- layer 1: gates = h0_t @ Wih1^T + h1_{t-1} @ Whh1^T -------------
        #pragma unroll
        for (int v = 0; v < 8; ++v) { acc0[v] = c1a; acc1[v] = c1b; }
        {
            v16h X0 = load_a_frag(h0s, lane, 0);
            v16h X1 = load_a_frag(h0s, lane, 1);
            v16h P0 = load_a_frag(h1s, lane, 0);
            v16h P1 = load_a_frag(h1s, lane, 1);
            acc0 = wmma_f16(X0, Bih1_a0, acc0);
            acc0 = wmma_f16(X1, Bih1_a1, acc0);
            acc0 = wmma_f16(P0, Bhh1_a0, acc0);
            acc0 = wmma_f16(P1, Bhh1_a1, acc0);
            acc1 = wmma_f16(X0, Bih1_b0, acc1);
            acc1 = wmma_f16(X1, Bih1_b1, acc1);
            acc1 = wmma_f16(P0, Bhh1_b0, acc1);
            acc1 = wmma_f16(P1, Bhh1_b1, acc1);
        }
        #pragma unroll
        for (int v = 0; v < 8; ++v) {
            gates[(v + sub * 8) * GS + n0] = acc0[v];
            gates[(v + sub * 8) * GS + n1] = acc1[v];
        }
        __syncthreads();

        // layer 1 elementwise LSTM cell
        #pragma unroll
        for (int q = 0; q < 4; ++q) {
            const int r = er0 + q;
            const float* g = gates + r * GS;
            float ig = fast_sigm(g[ej]);
            float fg = fast_sigm(g[64 + ej]);
            float gg = fast_tanh(g[128 + ej]);
            float og = fast_sigm(g[192 + ej]);
            float c  = fmaf(fg, cell1[q], ig * gg);
            cell1[q] = c;
            float h  = og * fast_tanh(c);
            h1s[r * HS + ej] = (_Float16)h;
            if (t == T_LEN - 1) hf[r * 64 + ej] = h;   // keep f32 for the FC head
        }
        __syncthreads();
    }

    // final FC: out[b] = h2_last[b,:] . Wfc + bfc
    if (tid < 16) {
        float s = bfc[0];
        #pragma unroll
        for (int j = 0; j < 64; ++j) s = fmaf(hf[tid * 64 + j], Wfc[j], s);
        out[bbase + tid] = s;
    }
}

extern "C" void kernel_launch(void* const* d_in, const int* in_sizes, int n_in,
                              void* d_out, int out_size, void* d_ws, size_t ws_size,
                              hipStream_t stream) {
    const float* x    = (const float*)d_in[0];
    const float* Wih0 = (const float*)d_in[1];
    const float* Whh0 = (const float*)d_in[2];
    const float* bih0 = (const float*)d_in[3];
    const float* bhh0 = (const float*)d_in[4];
    const float* Wih1 = (const float*)d_in[5];
    const float* Whh1 = (const float*)d_in[6];
    const float* bih1 = (const float*)d_in[7];
    const float* bhh1 = (const float*)d_in[8];
    const float* Wfc  = (const float*)d_in[9];
    const float* bfc  = (const float*)d_in[10];

    _Float16* w16 = (_Float16*)d_ws;    // 3 * 256*64 f16 = 96 KB scratch
    float* out = (float*)d_out;

    prep_weights_f16<<<64, 256, 0, stream>>>(Whh0, Wih1, Whh1, w16);
    lstm2_fused<<<4096 / 16, 256, 0, stream>>>(x, Wih0, bih0, bhh0, bih1, bhh1,
                                               Wfc, bfc, w16, out);
}
